// GIBModel_32246614459014
// MI455X (gfx1250) — compile-verified, matrix operand order checked by reading, not run on hardware
//
#include <hip/hip_runtime.h>
#include <hip/hip_bf16.h>

// ---------------------------------------------------------------------------
// GIB / GAT forward pass for MI455X (gfx1250, wave32).
//
// Roofline reasoning:
//  * Dominant op: x[50000,768] @ W1[768,512] = 39.3 GFLOP.  HBM traffic for x
//    is 153.6 MB -> ~6.6us at 23.3 TB/s.  fp32 WMMA (16x16x4) is ~8x slower
//    than the 16-bit matrix path, so we convert to bf16 while staging to LDS
//    and use v_wmma_f32_16x16x32_bf16 with fp32 accumulation, landing the
//    GEMM near the memory roofline.  (Pre-converting x to bf16 in HBM would
//    cost 307MB of traffic vs 154MB converting in-flight, so in-flight wins.)
//  * GEMM block tile 128x128, 8 waves each computing 32x64 (2x4 WMMA tiles,
//    8 wmma/wave/k-step), double-buffered LDS with one barrier per k-step and
//    global prefetch overlapped with the WMMA block.
//  * LDS row stride = 20 dwords: rows stay 16B aligned (ds_load_b128 fragment
//    gathers) and 20*m mod 64 banks is conflict-free across the 16 lanes.
//  * bf16 packing: round-half-up (+0x8000) + one v_perm_b32 byte-select =
//    3 VALU ops/pair (the two adds dual-issue), minimizing the VALU stream
//    that competes with WMMA/DS issue in the pipelined loop.  Exact ties
//    (the only divergence from RNE) occur with probability ~2^-16/element.
//  * GAT scatter phases are atomic/bandwidth bound -> wave-per-edge with
//    coalesced row reads and global_atomic_add_f32 accumulation.
// ---------------------------------------------------------------------------

typedef __attribute__((ext_vector_type(16))) __bf16 v16bf;
typedef __attribute__((ext_vector_type(8)))  float  v8f;

union FragBF { v16bf v; unsigned int u[8]; };

__device__ __forceinline__ unsigned int pack_bf2(float f0, float f1) {
  // round-half-up to bf16, pack two values with a single v_perm_b32
  unsigned int u0 = __float_as_uint(f0) + 0x8000u;
  unsigned int u1 = __float_as_uint(f1) + 0x8000u;
  return __builtin_amdgcn_perm(u1, u0, 0x07060302);  // {u1.b3,u1.b2,u0.b3,u0.b2}
}

// ---------------------------------------------------------------------------
// Zero fill (graph-capture safe memset replacement)
// ---------------------------------------------------------------------------
__global__ void fill_kernel(float* __restrict__ p, size_t n, float v) {
  size_t i = (size_t)blockIdx.x * blockDim.x + threadIdx.x;
  size_t stride = (size_t)gridDim.x * blockDim.x;
  for (; i < n; i += stride) p[i] = v;
}

// ---------------------------------------------------------------------------
// C[M,Nc] = act(A[M,K] @ B[K,Nc] (+bias)); A,B fp32 row-major, bf16 WMMA math.
// ---------------------------------------------------------------------------
#define TBM 128
#define TBN 128
#define TBK 32
#define LSTR (TBK / 2 + 4)   // 20 dwords: 16B-aligned rows, conflict-free

__global__ __launch_bounds__(256)
void gemm_bf16_wmma_kernel(const float* __restrict__ A, const float* __restrict__ B,
                           const float* __restrict__ bias, float* __restrict__ C,
                           int M, int K, int Nc, int do_relu) {
  __shared__ __align__(16) unsigned int lA[2][TBM][LSTR];
  __shared__ __align__(16) unsigned int lB[2][TBN][LSTR];

  const int tid  = threadIdx.x;
  const int lane = tid & 31;
  const int wv   = tid >> 5;
  const int bm   = blockIdx.x * TBM;
  const int bn   = blockIdx.y * TBN;
  const int wrow = (wv & 3) * 32;   // wave covers 32 rows
  const int wcol = (wv >> 2) * 64;  // wave covers 64 cols
  const int mrow = lane & 15;
  const int hh   = lane >> 4;

  v8f acc[2][4];
#pragma unroll
  for (int i = 0; i < 2; ++i)
#pragma unroll
    for (int j = 0; j < 4; ++j)
#pragma unroll
      for (int e = 0; e < 8; ++e) acc[i][j][e] = 0.0f;

  // Register staging for the double-buffered pipeline (pack to bf16 only at
  // LDS-store time so the vmem wait stays *after* the WMMA block).
  float2 ra[8], rb[8];

  auto loadTiles = [&](int k0) {
#pragma unroll
    for (int q = 0; q < 8; ++q) {
      int p = tid + q * 256;
      // A tile: 128 rows x 16 pairs; p = row*16 + kp (coalesced float2 reads)
      {
        int row = p >> 4, kp = p & 15;
        int gr = bm + row, gk = k0 + kp * 2;
        float2 f = make_float2(0.f, 0.f);
        if (gr < M) f = *(const float2*)&A[(size_t)gr * K + gk];
        ra[q] = f;
      }
      // B tile: 16 kpairs x 128 cols; p = kp*128 + n (coalesced over n)
      {
        int n = p & 127, kp = p >> 7;
        int gk = k0 + kp * 2, gn = bn + n;
        rb[q] = make_float2(B[(size_t)gk * Nc + gn], B[(size_t)(gk + 1) * Nc + gn]);
      }
    }
  };
  auto storeTiles = [&](int buf) {
#pragma unroll
    for (int q = 0; q < 8; ++q) {
      int p = tid + q * 256;
      lA[buf][p >> 4][p & 15]  = pack_bf2(ra[q].x, ra[q].y);
      lB[buf][p & 127][p >> 7] = pack_bf2(rb[q].x, rb[q].y);
    }
  };

  const int nk = K / TBK;
  loadTiles(0);
  storeTiles(0);

  for (int kt = 0; kt < nk; ++kt) {
    const int cur = kt & 1;
    __syncthreads();
    if (kt + 1 < nk) loadTiles((kt + 1) * TBK);   // prefetch overlaps WMMAs

    // Fragment gathers (CDNA5 16-bit layout): lane(0-15) M=lane,
    // VGPR v holds K pair (v<4 ? 0 : 16) + half*8 + (v&3)*2  ->  two b128 reads.
    FragBF a[2], b[4];
#pragma unroll
    for (int t = 0; t < 2; ++t) {
      int m = wrow + t * 16 + mrow;
      uint4 q0 = *(const uint4*)&lA[cur][m][hh * 4];
      uint4 q1 = *(const uint4*)&lA[cur][m][8 + hh * 4];
      a[t].u[0] = q0.x; a[t].u[1] = q0.y; a[t].u[2] = q0.z; a[t].u[3] = q0.w;
      a[t].u[4] = q1.x; a[t].u[5] = q1.y; a[t].u[6] = q1.z; a[t].u[7] = q1.w;
    }
#pragma unroll
    for (int t = 0; t < 4; ++t) {
      int n = wcol + t * 16 + mrow;
      uint4 q0 = *(const uint4*)&lB[cur][n][hh * 4];
      uint4 q1 = *(const uint4*)&lB[cur][n][8 + hh * 4];
      b[t].u[0] = q0.x; b[t].u[1] = q0.y; b[t].u[2] = q0.z; b[t].u[3] = q0.w;
      b[t].u[4] = q1.x; b[t].u[5] = q1.y; b[t].u[6] = q1.z; b[t].u[7] = q1.w;
    }
#pragma unroll
    for (int i = 0; i < 2; ++i)
#pragma unroll
      for (int j = 0; j < 4; ++j)
        acc[i][j] = __builtin_amdgcn_wmma_f32_16x16x32_bf16(
            false, a[i].v, false, b[j].v, (short)0, acc[i][j], false, false);

    if (kt + 1 < nk) storeTiles(cur ^ 1);   // other buffer: no extra barrier
  }

  // Epilogue. D layout: VGPR v -> M = v + 8*half, N = lane&15.
#pragma unroll
  for (int j = 0; j < 4; ++j) {
    int n = bn + wcol + j * 16 + mrow;
    float bj = bias ? bias[n] : 0.0f;
#pragma unroll
    for (int i = 0; i < 2; ++i)
#pragma unroll
      for (int v = 0; v < 8; ++v) {
        int m = bm + wrow + i * 16 + v + hh * 8;
        if (m < M) {
          float val = acc[i][j][v] + bj;
          if (do_relu) val = val > 0.f ? val : 0.f;
          C[(size_t)m * Nc + n] = val;
        }
      }
  }
}

// ---------------------------------------------------------------------------
// a_s[n,h] = <h[n,h,:], a_src[h,:]>,  a_d[n,h] = <h[n,h,:], a_dst[h,:]>
// wave per node, coalesced channel reads, shfl butterfly reduce.
// ---------------------------------------------------------------------------
__global__ __launch_bounds__(256)
void attn_scores_kernel(const float* __restrict__ h, const float* __restrict__ a_src,
                        const float* __restrict__ a_dst, float* __restrict__ as_o,
                        float* __restrict__ ad_o, int N, int H, int C) {
  int wid  = (int)(((size_t)blockIdx.x * blockDim.x + threadIdx.x) >> 5);
  int lane = threadIdx.x & 31;
  if (wid >= N) return;
  const float* hr = h + (size_t)wid * H * C;
  for (int head = 0; head < H; ++head) {
    float vs = 0.f, vd = 0.f;
    for (int c = lane; c < C; c += 32) {
      float hv = hr[head * C + c];
      vs += hv * a_src[head * C + c];
      vd += hv * a_dst[head * C + c];
    }
    for (int m = 16; m > 0; m >>= 1) {
      vs += __shfl_xor(vs, m, 32);
      vd += __shfl_xor(vd, m, 32);
    }
    if (lane == 0) { as_o[wid * H + head] = vs; ad_o[wid * H + head] = vd; }
  }
}

// ---------------------------------------------------------------------------
// Edge scatter: w = exp(leaky_relu(a_s[src]+a_d[dst])) (softmax shift-free),
// denom[dst] += w,  outacc[dst] += w * h[src].  Self loops appended in-kernel.
// ---------------------------------------------------------------------------
__global__ __launch_bounds__(256)
void edge_pass_kernel(const float* __restrict__ h, const float* __restrict__ as,
                      const float* __restrict__ ad, const int* __restrict__ esrc,
                      const int* __restrict__ edst, float* __restrict__ denom,
                      float* __restrict__ outacc, int N, int E, int H, int C) {
  int wid  = (int)(((size_t)blockIdx.x * blockDim.x + threadIdx.x) >> 5);
  int lane = threadIdx.x & 31;
  if (wid >= E + N) return;
  int s, d;
  if (wid < E) { s = esrc[wid]; d = edst[wid]; }
  else         { s = wid - E;   d = s; }

  float w[4];
  for (int head = 0; head < H; ++head) {
    float e = as[s * H + head] + ad[d * H + head];
    e = e > 0.f ? e : 0.2f * e;
    w[head] = __expf(e);
  }
  if (lane < H) atomicAdd(&denom[d * H + lane], w[lane]);

  const float* hs = h + (size_t)s * H * C;
  float*       od = outacc + (size_t)d * H * C;
  for (int idx = lane; idx < H * C; idx += 32)
    atomicAdd(&od[idx], w[idx / C] * hs[idx]);
}

// outacc[n,h,c] = relu(outacc/(denom+1e-16) + bias[h*C+c])   (in place)
__global__ void finalize_kernel(float* __restrict__ acc, const float* __restrict__ denom,
                                const float* __restrict__ bias, int N, int H, int C) {
  size_t i = (size_t)blockIdx.x * blockDim.x + threadIdx.x;
  size_t tot = (size_t)N * H * C;
  if (i >= tot) return;
  int n = (int)(i / (H * C));
  int rem = (int)(i % (H * C));
  int head = rem / C;
  float v = acc[i] / (denom[n * H + head] + 1e-16f) + bias[rem];
  acc[i] = v > 0.f ? v : 0.f;
}

// ---------------------------------------------------------------------------
// Per node: S = softmax(h @ Wclf + bclf); graph_sum += h; sub_sum += S0*h;
// counts += 1.  Wave per node, butterfly reduce gives all lanes the dots.
// ---------------------------------------------------------------------------
__global__ __launch_bounds__(256)
void node_clf_kernel(const float* __restrict__ h, const float* __restrict__ Wclf,
                     const float* __restrict__ bclf, const int* __restrict__ batch,
                     float* __restrict__ S, float* __restrict__ gsum,
                     float* __restrict__ ssum, float* __restrict__ counts, int N) {
  int wid  = (int)(((size_t)blockIdx.x * blockDim.x + threadIdx.x) >> 5);
  int lane = threadIdx.x & 31;
  if (wid >= N) return;
  const float* hr = h + (size_t)wid * 128;
  float hv[4];
  float s0 = 0.f, s1 = 0.f;
  for (int q = 0; q < 4; ++q) {
    int c = lane + q * 32;
    hv[q] = hr[c];
    s0 += hv[q] * Wclf[c * 2 + 0];
    s1 += hv[q] * Wclf[c * 2 + 1];
  }
  for (int m = 16; m > 0; m >>= 1) {
    s0 += __shfl_xor(s0, m, 32);
    s1 += __shfl_xor(s1, m, 32);
  }
  s0 += bclf[0]; s1 += bclf[1];
  float mx = fmaxf(s0, s1);
  float e0 = __expf(s0 - mx), e1 = __expf(s1 - mx);
  float inv = 1.0f / (e0 + e1);
  float S0 = e0 * inv, S1 = e1 * inv;
  int g = batch[wid];
  if (lane == 0) {
    S[wid * 2 + 0] = S0;
    S[wid * 2 + 1] = S1;
    atomicAdd(&counts[g], 1.0f);
  }
  for (int q = 0; q < 4; ++q) {
    int c = lane + q * 32;
    atomicAdd(&gsum[g * 128 + c], hv[q]);
    atomicAdd(&ssum[g * 128 + c], S0 * hv[q]);
  }
}

// agg[g] += S[src] (x) S[dst]  for same-graph edges (no self loops)
__global__ void edge_agg_kernel(const int* __restrict__ esrc, const int* __restrict__ edst,
                                const int* __restrict__ batch, const float* __restrict__ S,
                                float* __restrict__ agg, int E) {
  int e = blockIdx.x * blockDim.x + threadIdx.x;
  if (e >= E) return;
  int s = esrc[e], d = edst[e];
  int gs = batch[s];
  if (gs != batch[d]) return;
  float a0 = S[s * 2], a1 = S[s * 2 + 1];
  float b0 = S[d * 2], b1 = S[d * 2 + 1];
  atomicAdd(&agg[gs * 4 + 0], a0 * b0);
  atomicAdd(&agg[gs * 4 + 1], a0 * b1);
  atomicAdd(&agg[gs * 4 + 2], a1 * b0);
  atomicAdd(&agg[gs * 4 + 3], a1 * b1);
}

// graph_emb = gsum / max(counts,1)   (in place)
__global__ void graph_norm_kernel(float* __restrict__ gsum, const float* __restrict__ counts,
                                  int G, int C) {
  int i = blockIdx.x * blockDim.x + threadIdx.x;
  if (i >= G * C) return;
  gsum[i] = gsum[i] / fmaxf(counts[i / C], 1.0f);
}

// aggregate_loss = mean((diag(norm_adj)-1)^2); also writes detect_loss = 0
__global__ __launch_bounds__(128)
void agg_loss_kernel(const float* __restrict__ agg, float* __restrict__ out_scalar, int G) {
  __shared__ float red[128];
  int g = threadIdx.x;
  float a00 = agg[g * 4], a01 = agg[g * 4 + 1], a10 = agg[g * 4 + 2], a11 = agg[g * 4 + 3];
  float r0 = fabsf(a00) + fabsf(a01);
  float r1 = fabsf(a10) + fabsf(a11);
  float d0 = a00 / fmaxf(r0, 1e-5f);
  float d1 = a11 / fmaxf(r1, 1e-5f);
  red[g] = (d0 - 1.f) * (d0 - 1.f) + (d1 - 1.f) * (d1 - 1.f);
  __syncthreads();
  for (int s = 64; s > 0; s >>= 1) {
    if (g < s) red[g] += red[g + s];
    __syncthreads();
  }
  if (g == 0) {
    out_scalar[0] = red[0] / (2.0f * G);  // aggregate_loss
    out_scalar[1] = 0.0f;                 // detect_loss
  }
}

// MI estimator hidden layer for joint and margin inputs.
// NOTE: jax.random.permutation(key 42) is irreproducible without threefry;
// we substitute the fixed bijection g -> (37g+11) mod 128.
__global__ __launch_bounds__(128)
void mi_hidden_kernel(const float* __restrict__ gemb, const float* __restrict__ semb,
                      const float* __restrict__ Wib1, const float* __restrict__ bib1,
                      float* __restrict__ hidj, float* __restrict__ hidm) {
  int g = blockIdx.x, j = threadIdx.x;
  int gp = (g * 37 + 11) & 127;
  float aj = bib1[j], am = bib1[j];
  for (int k = 0; k < 128; ++k) {
    float w = Wib1[k * 128 + j];
    aj += gemb[g * 128 + k] * w;
    am += gemb[gp * 128 + k] * w;
  }
  for (int k = 0; k < 128; ++k) {
    float w = Wib1[(128 + k) * 128 + j];
    float sv = semb[g * 128 + k];
    aj += sv * w;
    am += sv * w;
  }
  hidj[g * 128 + j] = fmaxf(aj, 0.f);
  hidm[g * 128 + j] = fmaxf(am, 0.f);
}

// mi_loss = clip(log(clip(mean(exp(margin)),1,1e25)),-1e5,1e5) - mean(joint)
__global__ __launch_bounds__(128)
void mi_out_kernel(const float* __restrict__ hidj, const float* __restrict__ hidm,
                   const float* __restrict__ Wib2, const float* __restrict__ bib2,
                   float* __restrict__ out_mi, int G) {
  __shared__ float rj[128], rm[128];
  int g = threadIdx.x;
  float vj = bib2[0], vm = bib2[0];
  for (int k = 0; k < 128; ++k) {
    float w = Wib2[k];
    vj += hidj[g * 128 + k] * w;
    vm += hidm[g * 128 + k] * w;
  }
  rj[g] = fmaxf(vj, 0.f);
  rm[g] = __expf(fmaxf(vm, 0.f));
  __syncthreads();
  for (int s = 64; s > 0; s >>= 1) {
    if (g < s) { rj[g] += rj[g + s]; rm[g] += rm[g + s]; }
    __syncthreads();
  }
  if (g == 0) {
    float meanj = rj[0] / G;
    float meane = fminf(fmaxf(rm[0] / G, 1.0f), 1e25f);
    float mi = fminf(fmaxf(logf(meane), -1e5f), 1e5f);
    out_mi[0] = mi - meanj;
  }
}

// out[g] = relu(sub_emb[g] @ Wfc1 + bfc1) @ Wfc2 + bfc2
__global__ __launch_bounds__(128)
void fc_head_kernel(const float* __restrict__ semb, const float* __restrict__ Wfc1,
                    const float* __restrict__ bfc1, const float* __restrict__ Wfc2,
                    const float* __restrict__ bfc2, float* __restrict__ out) {
  __shared__ float hid[128];
  int g = blockIdx.x, t = threadIdx.x;
  float a = bfc1[t];
  for (int k = 0; k < 128; ++k) a += semb[g * 128 + k] * Wfc1[k * 128 + t];
  hid[t] = fmaxf(a, 0.f);
  __syncthreads();
  if (t < 2) {
    float o = bfc2[t];
    for (int k = 0; k < 128; ++k) o += hid[k] * Wfc2[k * 2 + t];
    out[g * 2 + t] = o;
  }
}

// ---------------------------------------------------------------------------
// Host orchestration
// ---------------------------------------------------------------------------
static inline void fill0(float* p, size_t n, hipStream_t s) {
  int blocks = (int)((n + 255) / 256);
  if (blocks > 4096) blocks = 4096;
  fill_kernel<<<blocks, 256, 0, s>>>(p, n, 0.0f);
}

extern "C" void kernel_launch(void* const* d_in, const int* in_sizes, int n_in,
                              void* d_out, int out_size, void* d_ws, size_t ws_size,
                              hipStream_t stream) {
  const float* x      = (const float*)d_in[0];
  const int*   edge   = (const int*)  d_in[1];   // [2, E]
  const int*   batch  = (const int*)  d_in[2];
  const float* W1     = (const float*)d_in[3];
  const float* a_src1 = (const float*)d_in[4];
  const float* a_dst1 = (const float*)d_in[5];
  const float* b1     = (const float*)d_in[6];
  const float* W2     = (const float*)d_in[7];
  const float* a_src2 = (const float*)d_in[8];
  const float* a_dst2 = (const float*)d_in[9];
  const float* b2     = (const float*)d_in[10];
  const float* Wclf   = (const float*)d_in[11];
  const float* bclf   = (const float*)d_in[12];
  const float* Wfc1   = (const float*)d_in[13];
  const float* bfc1   = (const float*)d_in[14];
  const float* Wfc2   = (const float*)d_in[15];
  const float* bfc2   = (const float*)d_in[16];
  const float* Wib1   = (const float*)d_in[17];
  const float* bib1   = (const float*)d_in[18];
  const float* Wib2   = (const float*)d_in[19];
  const float* bib2   = (const float*)d_in[20];

  const int F = 768, HID = 128, HEADS = 4, G = 128;
  const int N = in_sizes[0] / F;
  const int E = in_sizes[1] / 2;
  const int* esrc = edge;
  const int* edst = edge + E;

  // Workspace layout (floats). buf2 is reused: GAT1 accum -> h1relu -> GAT2
  // accum -> h_final.  buf1 is reused: h1 -> h2 (first N*128).
  float* ws    = (float*)d_ws;
  float* buf1  = ws;                               // N*512
  float* buf2  = buf1 + (size_t)N * 512;           // N*512
  float* ext   = buf2 + (size_t)N * 512;
  float* as1   = ext;  ext += (size_t)N * HEADS;
  float* ad1   = ext;  ext += (size_t)N * HEADS;
  float* den1  = ext;  ext += (size_t)N * HEADS;
  float* as2   = ext;  ext += N;
  float* ad2   = ext;  ext += N;
  float* den2  = ext;  ext += N;
  float* S     = ext;  ext += (size_t)N * 2;
  float* gsum  = ext;  ext += G * HID;
  float* ssum  = ext;  ext += G * HID;
  float* cnts  = ext;  ext += G;
  float* agg   = ext;  ext += G * 4;
  float* hidj  = ext;  ext += G * HID;
  float* hidm  = ext;  ext += G * HID;

  float* out      = (float*)d_out;        // [G,2] flat
  float* out_scal = out + 2 * G;          // [aggregate, detect, mi]

  const int nodeWaveBlocks = (N + 7) / 8;           // wave-per-node, 8 waves/block
  const int edgeWaveBlocks = (E + N + 7) / 8;       // wave-per-edge(+self loop)

  // ---- GAT layer 1 -------------------------------------------------------
  {
    dim3 grid((N + TBM - 1) / TBM, (HEADS * HID) / TBN);
    gemm_bf16_wmma_kernel<<<grid, 256, 0, stream>>>(x, W1, nullptr, buf1,
                                                    N, F, HEADS * HID, 0);
  }
  attn_scores_kernel<<<nodeWaveBlocks, 256, 0, stream>>>(buf1, a_src1, a_dst1,
                                                         as1, ad1, N, HEADS, HID);
  fill0(den1, (size_t)N * HEADS, stream);
  fill0(buf2, (size_t)N * 512, stream);
  edge_pass_kernel<<<edgeWaveBlocks, 256, 0, stream>>>(buf1, as1, ad1, esrc, edst,
                                                       den1, buf2, N, E, HEADS, HID);
  {
    size_t tot = (size_t)N * 512;
    finalize_kernel<<<(int)((tot + 255) / 256), 256, 0, stream>>>(buf2, den1, b1,
                                                                  N, HEADS, HID);
  }

  // ---- GAT layer 2 (h2 into buf1; accum back into buf2) ------------------
  {
    dim3 grid((N + TBM - 1) / TBM, HID / TBN);
    gemm_bf16_wmma_kernel<<<grid, 256, 0, stream>>>(buf2, W2, nullptr, buf1,
                                                    N, HEADS * HID, HID, 0);
  }
  attn_scores_kernel<<<nodeWaveBlocks, 256, 0, stream>>>(buf1, a_src2, a_dst2,
                                                         as2, ad2, N, 1, HID);
  fill0(den2, (size_t)N, stream);
  fill0(buf2, (size_t)N * HID, stream);
  edge_pass_kernel<<<edgeWaveBlocks, 256, 0, stream>>>(buf1, as2, ad2, esrc, edst,
                                                       den2, buf2, N, E, 1, HID);
  {
    size_t tot = (size_t)N * HID;
    finalize_kernel<<<(int)((tot + 255) / 256), 256, 0, stream>>>(buf2, den2, b2,
                                                                  N, 1, HID);
  }

  // ---- Subgraph classifier + graph pooling -------------------------------
  fill0(gsum, (size_t)G * HID, stream);
  fill0(ssum, (size_t)G * HID, stream);
  fill0(cnts, (size_t)G, stream);
  fill0(agg, (size_t)G * 4, stream);
  node_clf_kernel<<<nodeWaveBlocks, 256, 0, stream>>>(buf2, Wclf, bclf, batch,
                                                      S, gsum, ssum, cnts, N);
  edge_agg_kernel<<<(E + 255) / 256, 256, 0, stream>>>(esrc, edst, batch, S, agg, E);
  graph_norm_kernel<<<(G * HID + 255) / 256, 256, 0, stream>>>(gsum, cnts, G, HID);

  // ---- Losses + head -----------------------------------------------------
  agg_loss_kernel<<<1, 128, 0, stream>>>(agg, out_scal, G);
  mi_hidden_kernel<<<G, 128, 0, stream>>>(gsum, ssum, Wib1, bib1, hidj, hidm);
  mi_out_kernel<<<1, 128, 0, stream>>>(hidj, hidm, Wib2, bib2, out_scal + 2, G);
  fc_head_kernel<<<G, 128, 0, stream>>>(ssum, Wfc1, bfc1, Wfc2, bfc2, out);

  (void)n_in; (void)out_size; (void)ws_size;
}
